// Processor_7138235646193
// MI455X (gfx1250) — matrix operand order repaired and checked by reading, not compiled, not run on hardware
//
#include <hip/hip_runtime.h>
#include <hip/hip_bf16.h>

// ---------------------------------------------------------------------------
// MetaLayer GNN (L=4) on gfx1250: f16 WMMA (v_wmma_f32_16x16x32_f16) with
// fp32 accumulation. Edge MLP dominates (39 GFLOP/layer); with f16 WMMA the
// layer is HBM-bound (~0.5 GB/layer @ 23.3 TB/s), so f16 is the right
// precision. Weights are pre-transposed to [n][k] f16 so B fragments are two
// contiguous 16B loads; activations are kept in f16 shadow buffers so A
// fragments load straight from global.
//
// WMMA f16 16x16x32 register layouts assumed (CDNA5 ISA 7.12.2):
//   A (16x32): lane = (h= lane>>4, r = lane&15), row M=r,
//              elems 0..7  -> K = kbase + 8h + e
//              elems 8..15 -> K = kbase + 16 + 8h + (e-8)
//   B (32x16): lane col N=r, same K mapping as A
//   C/D (16x16 f32): elem v -> M = v + 8h, N = r
// ---------------------------------------------------------------------------

#define F_DIM   128
#define N_NODES 50000
#define E_EDGES 600000
#define B_GR    64
#define L_LAY   4
#define LDH     136   // padded LDS row stride (halfs): 272B, 16B aligned, conflict-free

typedef __attribute__((ext_vector_type(16))) _Float16 h16;
typedef __attribute__((ext_vector_type(8)))  _Float16 h8;
typedef __attribute__((ext_vector_type(8)))  float    f8v;

__device__ __forceinline__ f8v wmma_f16(h16 a, h16 b, f8v c) {
  return __builtin_amdgcn_wmma_f32_16x16x32_f16(false, a, false, b, (short)0, c,
                                                false, false);
}

// A fragment: rowbase points at row M=r (contiguous K). Two 16B loads.
__device__ __forceinline__ h16 frag_a(const _Float16* rowbase, int kbase, int h) {
  h8 lo = *(const h8*)(rowbase + kbase + h * 8);
  h8 hi = *(const h8*)(rowbase + kbase + 16 + h * 8);
  return __builtin_shufflevector(lo, hi, 0,1,2,3,4,5,6,7,8,9,10,11,12,13,14,15);
}
// B fragment: colbase points at W^T row n (contiguous K). Two 16B loads.
__device__ __forceinline__ h16 frag_b(const _Float16* colbase, int kbase, int h) {
  h8 lo = *(const h8*)(colbase + kbase + h * 8);
  h8 hi = *(const h8*)(colbase + kbase + 16 + h * 8);
  return __builtin_shufflevector(lo, hi, 0,1,2,3,4,5,6,7,8,9,10,11,12,13,14,15);
}

__device__ __forceinline__ void lds_wait() {
  asm volatile("s_wait_dscnt 0" ::: "memory");
}

// ------------------------------ utility kernels ----------------------------

__global__ void zero_kernel(float* __restrict__ p, int n) {
  int i = blockIdx.x * 256 + threadIdx.x;
  if (i < n) p[i] = 0.0f;
}

__global__ void f32_to_f16_kernel(const float* __restrict__ s,
                                  _Float16* __restrict__ d, int n) {
  int i = blockIdx.x * 256 + threadIdx.x;
  if (i < n) d[i] = (_Float16)s[i];
}

// W [K][Nn] f32 row-major -> W^T [Nn][K] f16
__global__ void wtrans_kernel(const float* __restrict__ src,
                              _Float16* __restrict__ dst, int K, int Nn) {
  int i = blockIdx.x * 256 + threadIdx.x;
  if (i < K * Nn) {
    int k = i / Nn, n = i - k * Nn;
    dst[n * K + k] = (_Float16)src[i];
  }
}

__global__ void deg_kernel(const int* __restrict__ col, float* __restrict__ deg) {
  int e = blockIdx.x * 256 + threadIdx.x;
  if (e < E_EDGES) atomicAdd(deg + col[e], 1.0f);
}

__global__ void bcnt_kernel(const int* __restrict__ batch, float* __restrict__ bc) {
  int i = blockIdx.x * 256 + threadIdx.x;
  if (i < N_NODES) atomicAdd(bc + batch[i], 1.0f);
}

// hcat[n] = [ xh[n], agg[n]/max(deg,1), uh[batch[n]] ]   (f16, [N][384])
__global__ void prep_node_kernel(const _Float16* __restrict__ xh,
                                 const float* __restrict__ agg,
                                 const float* __restrict__ deg,
                                 const _Float16* __restrict__ uh,
                                 const int* __restrict__ batch,
                                 _Float16* __restrict__ hcat) {
  int i = blockIdx.x * 256 + threadIdx.x;
  if (i < N_NODES * 128) {
    int n = i >> 7, c = i & 127;
    float rd = 1.0f / fmaxf(deg[n], 1.0f);
    _Float16* drow = hcat + (size_t)n * 384;
    drow[c]       = xh[i];
    drow[128 + c] = (_Float16)(agg[i] * rd);
    drow[256 + c] = uh[batch[n] * 128 + c];
  }
}

// gh[b] = [ uh[b], xsum[b]/max(bcnt,1) ]   (f16, [64][256])
__global__ void prep_global_kernel(const _Float16* __restrict__ uh,
                                   const float* __restrict__ xsum,
                                   const float* __restrict__ bcnt,
                                   _Float16* __restrict__ gh) {
  int i = blockIdx.x * 256 + threadIdx.x;
  if (i < B_GR * 128) {
    int b = i >> 7, c = i & 127;
    gh[b * 256 + c]       = uh[i];
    gh[b * 256 + 128 + c] = (_Float16)(xsum[i] / fmaxf(bcnt[b], 1.0f));
  }
}

// ------------------------------ edge MLP -----------------------------------
// m = relu(x[row] @ W1 + b1) @ W2 + b2 ; atomic scatter-add into agg[col].
// 8 waves/block, 16 edges per wave, 128 edges/block.
__global__ __launch_bounds__(256) void edge_mlp_kernel(
    const _Float16* __restrict__ xh,
    const int* __restrict__ row_idx, const int* __restrict__ col_idx,
    const _Float16* __restrict__ w1t, const float* __restrict__ b1,
    const _Float16* __restrict__ w2t, const float* __restrict__ b2,
    float* __restrict__ agg) {
  __shared__ _Float16 Hs[8 * 16 * LDH];
  const int t = threadIdx.x;
  const int wave = t >> 5, lane = t & 31;
  const int h = lane >> 4, r = lane & 15;
  const int ebase = blockIdx.x * 128 + wave * 16;
  if (ebase >= E_EDGES) return;   // E % 16 == 0 -> tiles always full
  _Float16* Hw = Hs + wave * 16 * LDH;

  const int myrow = row_idx[ebase + r];
  const _Float16* xrow = xh + (size_t)myrow * 128;

  h16 afr[4];
#pragma unroll
  for (int kc = 0; kc < 4; ++kc) afr[kc] = frag_a(xrow, kc * 32, h);

  // GEMM1 + ReLU -> LDS
#pragma unroll
  for (int nt = 0; nt < 8; ++nt) {
    const int n = nt * 16 + r;
    const float bv = b1[n];
    f8v c;
#pragma unroll
    for (int v = 0; v < 8; ++v) c[v] = bv;
    const _Float16* wcol = w1t + n * 128;
#pragma unroll
    for (int kc = 0; kc < 4; ++kc)
      c = wmma_f16(afr[kc], frag_b(wcol, kc * 32, h), c);
#pragma unroll
    for (int v = 0; v < 8; ++v)
      Hw[(v + 8 * h) * LDH + n] = (_Float16)fmaxf(c[v], 0.0f);
  }
  lds_wait();

  h16 hfr[4];
#pragma unroll
  for (int kc = 0; kc < 4; ++kc) hfr[kc] = frag_a(Hw + r * LDH, kc * 32, h);

  // GEMM2 + scatter-mean numerator
#pragma unroll
  for (int nt = 0; nt < 8; ++nt) {
    const int n = nt * 16 + r;
    const float bv = b2[n];
    f8v c;
#pragma unroll
    for (int v = 0; v < 8; ++v) c[v] = bv;
    const _Float16* wcol = w2t + n * 128;
#pragma unroll
    for (int kc = 0; kc < 4; ++kc)
      c = wmma_f16(hfr[kc], frag_b(wcol, kc * 32, h), c);
#pragma unroll
    for (int v = 0; v < 8; ++v) {
      const int dst = col_idx[ebase + v + 8 * h];
      atomicAdd(agg + (size_t)dst * 128 + n, c[v]);
    }
  }
}

// ------------------------------ node MLP -----------------------------------
// x' = relu(hcat @ W1 + b1) @ W2 + b2 ; also accumulate per-graph sums.
__global__ __launch_bounds__(256) void node_mlp_kernel(
    const _Float16* __restrict__ hcat, const int* __restrict__ batch,
    const _Float16* __restrict__ w1t, const float* __restrict__ b1,
    const _Float16* __restrict__ w2t, const float* __restrict__ b2,
    float* __restrict__ xdst, _Float16* __restrict__ xh_out,
    float* __restrict__ xsum) {
  __shared__ _Float16 Hs[8 * 16 * LDH];
  const int t = threadIdx.x;
  const int wave = t >> 5, lane = t & 31;
  const int h = lane >> 4, r = lane & 15;
  const int nbase = blockIdx.x * 128 + wave * 16;
  if (nbase >= N_NODES) return;   // N % 16 == 0
  _Float16* Hw = Hs + wave * 16 * LDH;

  const _Float16* arow = hcat + (size_t)(nbase + r) * 384;
  h16 afr[12];
#pragma unroll
  for (int kc = 0; kc < 12; ++kc) afr[kc] = frag_a(arow, kc * 32, h);

#pragma unroll
  for (int nt = 0; nt < 8; ++nt) {
    const int n = nt * 16 + r;
    const float bv = b1[n];
    f8v c;
#pragma unroll
    for (int v = 0; v < 8; ++v) c[v] = bv;
    const _Float16* wcol = w1t + n * 384;
#pragma unroll
    for (int kc = 0; kc < 12; ++kc)
      c = wmma_f16(afr[kc], frag_b(wcol, kc * 32, h), c);
#pragma unroll
    for (int v = 0; v < 8; ++v)
      Hw[(v + 8 * h) * LDH + n] = (_Float16)fmaxf(c[v], 0.0f);
  }
  lds_wait();

  h16 hfr[4];
#pragma unroll
  for (int kc = 0; kc < 4; ++kc) hfr[kc] = frag_a(Hw + r * LDH, kc * 32, h);

#pragma unroll
  for (int nt = 0; nt < 8; ++nt) {
    const int n = nt * 16 + r;
    const float bv = b2[n];
    f8v c;
#pragma unroll
    for (int v = 0; v < 8; ++v) c[v] = bv;
    const _Float16* wcol = w2t + n * 128;
#pragma unroll
    for (int kc = 0; kc < 4; ++kc)
      c = wmma_f16(hfr[kc], frag_b(wcol, kc * 32, h), c);
#pragma unroll
    for (int v = 0; v < 8; ++v) {
      const int i = nbase + v + 8 * h;
      const float val = c[v];
      xdst[(size_t)i * 128 + n] = val;
      xh_out[(size_t)i * 128 + n] = (_Float16)val;
      atomicAdd(xsum + batch[i] * 128 + n, val);
    }
  }
}

// ------------------------------ global MLP ---------------------------------
// u' = relu(gh @ W1 + b1) @ W2 + b2 ; one block, 4 waves (B=64 -> 4 tiles)
__global__ __launch_bounds__(128) void global_mlp_kernel(
    const _Float16* __restrict__ gh,
    const _Float16* __restrict__ w1t, const float* __restrict__ b1,
    const _Float16* __restrict__ w2t, const float* __restrict__ b2,
    float* __restrict__ udst, _Float16* __restrict__ uh_out) {
  __shared__ _Float16 Hs[4 * 16 * LDH];
  const int t = threadIdx.x;
  const int wave = t >> 5, lane = t & 31;
  const int h = lane >> 4, r = lane & 15;
  const int gbase = wave * 16;
  _Float16* Hw = Hs + wave * 16 * LDH;

  const _Float16* arow = gh + (size_t)(gbase + r) * 256;
  h16 afr[8];
#pragma unroll
  for (int kc = 0; kc < 8; ++kc) afr[kc] = frag_a(arow, kc * 32, h);

#pragma unroll
  for (int nt = 0; nt < 8; ++nt) {
    const int n = nt * 16 + r;
    const float bv = b1[n];
    f8v c;
#pragma unroll
    for (int v = 0; v < 8; ++v) c[v] = bv;
    const _Float16* wcol = w1t + n * 256;
#pragma unroll
    for (int kc = 0; kc < 8; ++kc)
      c = wmma_f16(afr[kc], frag_b(wcol, kc * 32, h), c);
#pragma unroll
    for (int v = 0; v < 8; ++v)
      Hw[(v + 8 * h) * LDH + n] = (_Float16)fmaxf(c[v], 0.0f);
  }
  lds_wait();

  h16 hfr[4];
#pragma unroll
  for (int kc = 0; kc < 4; ++kc) hfr[kc] = frag_a(Hw + r * LDH, kc * 32, h);

#pragma unroll
  for (int nt = 0; nt < 8; ++nt) {
    const int n = nt * 16 + r;
    const float bv = b2[n];
    f8v c;
#pragma unroll
    for (int v = 0; v < 8; ++v) c[v] = bv;
    const _Float16* wcol = w2t + n * 128;
#pragma unroll
    for (int kc = 0; kc < 4; ++kc)
      c = wmma_f16(hfr[kc], frag_b(wcol, kc * 32, h), c);
#pragma unroll
    for (int v = 0; v < 8; ++v) {
      const int b = gbase + v + 8 * h;
      udst[b * 128 + n] = c[v];
      uh_out[b * 128 + n] = (_Float16)c[v];
    }
  }
}

// ------------------------------ host orchestration -------------------------

extern "C" void kernel_launch(void* const* d_in, const int* in_sizes, int n_in,
                              void* d_out, int out_size, void* d_ws, size_t ws_size,
                              hipStream_t stream) {
  (void)in_sizes; (void)n_in; (void)out_size; (void)ws_size;
  const float* x_in  = (const float*)d_in[0];
  const int*   eidx  = (const int*)d_in[1];     // [2][E]
  const float* u_in  = (const float*)d_in[2];
  const int*   batch = (const int*)d_in[3];
  const float* n1w1 = (const float*)d_in[4];
  const float* n1b1 = (const float*)d_in[5];
  const float* n1w2 = (const float*)d_in[6];
  const float* n1b2 = (const float*)d_in[7];
  const float* n2w1 = (const float*)d_in[8];
  const float* n2b1 = (const float*)d_in[9];
  const float* n2w2 = (const float*)d_in[10];
  const float* n2b2 = (const float*)d_in[11];
  const float* gw1  = (const float*)d_in[12];
  const float* gb1  = (const float*)d_in[13];
  const float* gw2  = (const float*)d_in[14];
  const float* gb2  = (const float*)d_in[15];
  float* out = (float*)d_out;

  // ---- workspace carve ----
  char* w = (char*)d_ws;
  auto alloc = [&](size_t bytes) -> char* {
    char* p = w; w += (bytes + 255) & ~(size_t)255; return p;
  };
  float*    agg  = (float*)alloc((size_t)N_NODES * 128 * 4);
  float*    xsum = (float*)alloc((size_t)B_GR * 128 * 4);
  float*    deg  = (float*)alloc((size_t)N_NODES * 4);
  float*    bcnt = (float*)alloc((size_t)B_GR * 4);
  float*    xA   = (float*)alloc((size_t)N_NODES * 128 * 4);
  float*    uA   = (float*)alloc((size_t)B_GR * 128 * 4);
  _Float16* xh   = (_Float16*)alloc((size_t)N_NODES * 128 * 2);
  _Float16* uh   = (_Float16*)alloc((size_t)B_GR * 128 * 2);
  _Float16* hcat = (_Float16*)alloc((size_t)N_NODES * 384 * 2);
  _Float16* gh   = (_Float16*)alloc((size_t)B_GR * 256 * 2);
  _Float16* n1w1t = (_Float16*)alloc((size_t)L_LAY * 128 * 128 * 2);
  _Float16* n1w2t = (_Float16*)alloc((size_t)L_LAY * 128 * 128 * 2);
  _Float16* n2w1t = (_Float16*)alloc((size_t)L_LAY * 384 * 128 * 2);
  _Float16* n2w2t = (_Float16*)alloc((size_t)L_LAY * 128 * 128 * 2);
  _Float16* gw1t  = (_Float16*)alloc((size_t)L_LAY * 256 * 128 * 2);
  _Float16* gw2t  = (_Float16*)alloc((size_t)L_LAY * 128 * 128 * 2);

  auto cdiv = [](int a, int b) { return (a + b - 1) / b; };

  // ---- one-time per launch: counts + f16 weight transposes + f16 activations
  zero_kernel<<<cdiv(N_NODES, 256), 256, 0, stream>>>(deg, N_NODES);
  zero_kernel<<<1, 256, 0, stream>>>(bcnt, B_GR);
  deg_kernel<<<cdiv(E_EDGES, 256), 256, 0, stream>>>(eidx + E_EDGES, deg);
  bcnt_kernel<<<cdiv(N_NODES, 256), 256, 0, stream>>>(batch, bcnt);

  for (int l = 0; l < L_LAY; ++l) {
    wtrans_kernel<<<cdiv(128 * 128, 256), 256, 0, stream>>>(
        n1w1 + (size_t)l * 128 * 128, n1w1t + (size_t)l * 128 * 128, 128, 128);
    wtrans_kernel<<<cdiv(128 * 128, 256), 256, 0, stream>>>(
        n1w2 + (size_t)l * 128 * 128, n1w2t + (size_t)l * 128 * 128, 128, 128);
    wtrans_kernel<<<cdiv(384 * 128, 256), 256, 0, stream>>>(
        n2w1 + (size_t)l * 384 * 128, n2w1t + (size_t)l * 384 * 128, 384, 128);
    wtrans_kernel<<<cdiv(128 * 128, 256), 256, 0, stream>>>(
        n2w2 + (size_t)l * 128 * 128, n2w2t + (size_t)l * 128 * 128, 128, 128);
    wtrans_kernel<<<cdiv(256 * 128, 256), 256, 0, stream>>>(
        gw1 + (size_t)l * 256 * 128, gw1t + (size_t)l * 256 * 128, 256, 128);
    wtrans_kernel<<<cdiv(128 * 128, 256), 256, 0, stream>>>(
        gw2 + (size_t)l * 128 * 128, gw2t + (size_t)l * 128 * 128, 128, 128);
  }
  f32_to_f16_kernel<<<cdiv(N_NODES * 128, 256), 256, 0, stream>>>(x_in, xh, N_NODES * 128);
  f32_to_f16_kernel<<<cdiv(B_GR * 128, 256), 256, 0, stream>>>(u_in, uh, B_GR * 128);

  // ---- layer loop ----
  for (int l = 0; l < L_LAY; ++l) {
    float* xdst = (l == L_LAY - 1) ? out : xA;
    float* udst = (l == L_LAY - 1) ? out + (size_t)N_NODES * 128 : uA;

    zero_kernel<<<cdiv(N_NODES * 128, 256), 256, 0, stream>>>(agg, N_NODES * 128);
    zero_kernel<<<cdiv(B_GR * 128, 256), 256, 0, stream>>>(xsum, B_GR * 128);

    edge_mlp_kernel<<<cdiv(E_EDGES, 128), 256, 0, stream>>>(
        xh, eidx, eidx + E_EDGES,
        n1w1t + (size_t)l * 128 * 128, n1b1 + (size_t)l * 128,
        n1w2t + (size_t)l * 128 * 128, n1b2 + (size_t)l * 128, agg);

    prep_node_kernel<<<cdiv(N_NODES * 128, 256), 256, 0, stream>>>(
        xh, agg, deg, uh, batch, hcat);

    node_mlp_kernel<<<cdiv(N_NODES, 128), 256, 0, stream>>>(
        hcat, batch,
        n2w1t + (size_t)l * 384 * 128, n2b1 + (size_t)l * 128,
        n2w2t + (size_t)l * 128 * 128, n2b2 + (size_t)l * 128,
        xdst, xh, xsum);

    prep_global_kernel<<<cdiv(B_GR * 128, 256), 256, 0, stream>>>(uh, xsum, bcnt, gh);

    global_mlp_kernel<<<1, 128, 0, stream>>>(
        gh,
        gw1t + (size_t)l * 256 * 128, gb1 + (size_t)l * 128,
        gw2t + (size_t)l * 128 * 128, gb2 + (size_t)l * 128,
        udst, uh);
  }
}
